// Attention_34230889349279
// MI455X (gfx1250) — matrix operand order repaired and checked by reading, not compiled, not run on hardware
//
#include <hip/hip_runtime.h>
#include <hip/hip_bf16.h>
#include <stdint.h>

// ---------------------------------------------------------------------------
// CDNA5 (gfx1250) attention kernel:
//   - TDM tensor_load_to_lds, double-buffered: item ib+1's 4D patch-gather
//     DMA runs while item ib is computed (TENSORcnt in-order => wait(1)).
//   - bf16 WMMA (v_wmma_f32_16x16x32_bf16) for the C->hidden projection GEMM
//   - fp32 VALU for the tiny 9x9 attention / softmax / attn@v stages
// One workgroup (256 thr = 8 wave32) per batch item, BPW items per WG.
// ---------------------------------------------------------------------------

typedef __bf16 bf16;
typedef __bf16 v16bf __attribute__((ext_vector_type(16)));
typedef __bf16 v8bf  __attribute__((ext_vector_type(8)));
typedef float  v8f   __attribute__((ext_vector_type(8)));
typedef unsigned int u32x4 __attribute__((ext_vector_type(4)));
typedef int    i32x8 __attribute__((ext_vector_type(8)));
typedef int    i32x4 __attribute__((ext_vector_type(4)));

#define BPW 8              // batch items per workgroup
#define NTOK 9             // attention tokens
#define NPIX 9             // pixels per token
#define CCH  256           // channels
#define HID  128           // hidden
#define ROWS 81            // NTOK*NPIX rows of the projection GEMM
#define ROWS_PAD 96        // padded to 6 m-tiles of 16
#define ELEMS 20736        // 81*256 floats per batch item

#if __has_builtin(__builtin_amdgcn_tensor_load_to_lds)
#define HAVE_TDM 1
#else
#define HAVE_TDM 0
#endif

#if HAVE_TDM
// Issue one 4D TDM gather of x[b] (81x256 f32) into LDS at lds_off:
// dim0 = 768 contiguous floats (s4,c); dim1 = s2 (stride 2304);
// dim2 = s3 (stride 768); dim3 = s1 (stride 6912).
__device__ __forceinline__ void issue_tdm(uint32_t lds_off, const float* xb)
{
    uint64_t ga = (uint64_t)(uintptr_t)xb;
    u32x4 g0 = { 1u,                                   // count=1, load
                 lds_off,                              // lds_addr
                 (uint32_t)ga,                         // global_addr lo
                 (uint32_t)((ga >> 32) & 0x1FFFFFFu) | (2u << 30) }; // hi | type=2
    i32x8 g1 = { 0x00020000,        // data_size=4B
                 0x03000000,        // tensor_dim0 = 768 (lo16 in [31:16])
                 0x00030000,        // tensor_dim0 hi / tensor_dim1 = 3
                 0x03000000,        // tensor_dim1 hi / tile_dim0 = 768
                 0x00030003,        // tile_dim1 = 3, tile_dim2 = 3
                 2304,              // tensor_dim0_stride lo32
                 0x03000000,        // dim0_stride hi / dim1_stride = 768 lo16
                 0 };               // dim1_stride hi
    i32x4 g2 = { 3,                 // tensor_dim2 = 3
                 3,                 // tensor_dim3 = 3
                 6912,              // tensor_dim2_stride lo32
                 0x00030000 };      // dim2_stride hi / tile_dim3 = 3
    i32x4 g3 = { 0, 0, 0, 0 };
    i32x8 g4 = { 0, 0, 0, 0, 0, 0, 0, 0 };  // reserved group (clang-23 arity)
    __builtin_amdgcn_tensor_load_to_lds(g0, g1, g2, g3, g4, 0);
}
#endif

__global__ void __launch_bounds__(256, 1)
attn_pix_kernel(const float* __restrict__ x,
                const float* __restrict__ W_fc,
                const float* __restrict__ b_fc,
                const int*   __restrict__ block_idx,
                const int*   __restrict__ match_vec,
                float* __restrict__ out,
                int Btot)
{
    __shared__ __align__(16) float xfc[2][ROWS * CCH];    // double-buffered TDM dest 162 KB
    __shared__ __align__(16) bf16  abf[ROWS_PAD * CCH];   // bf16 A matrix             48 KB
    __shared__ __align__(16) float qld[ROWS * HID];       // q                         40.5 KB
    __shared__ __align__(16) float attn_l[NTOK * NTOK];   // logits/probs

    const int tid  = threadIdx.x;
    const int wave = tid >> 5;      // 0..7  == n-tile of the GEMM
    const int lane = tid & 31;
    const int lhal = lane >> 4;     // 0/1 lane half
    const int lmod = lane & 15;

    const int b0 = blockIdx.x * BPW;

#if HAVE_TDM
    // Prefetch item 0 before the W preload so the DMA overlaps global reads.
    if (__builtin_amdgcn_readfirstlane(wave) == 0 && b0 < Btot) {
        issue_tdm((uint32_t)(uintptr_t)&xfc[0][0], x + (size_t)b0 * ELEMS);
    }
#endif

    // ------------------------------------------------------------------
    // Preload this wave's 8 B fragments (W_fc^T, 32x16 bf16 each) into
    // registers once.  B layout: v16bf element e = B[kbase + lhal*16 + e,
    // n = lmod] = W_fc[wave*16+lmod][kbase + lhal*16 + e]  -> 16
    // contiguous floats per lane per k-step, convert f32->bf16 in regs.
    // ------------------------------------------------------------------
    v16bf bfrag[8];
    {
        const float* wrow = W_fc + (size_t)(wave * 16 + lmod) * CCH;
#pragma unroll
        for (int kk = 0; kk < 8; ++kk) {
            const float* src = wrow + kk * 32 + lhal * 16;
            v16bf bv;
#pragma unroll
            for (int e = 0; e < 16; ++e) bv[e] = (bf16)src[e];
            bfrag[kk] = bv;
        }
    }
    const float bias = b_fc[wave * 16 + lmod];

    for (int ib = 0; ib < BPW; ++ib) {
        const int b = b0 + ib;
        if (b >= Btot) break;
        float* cur = &xfc[ib & 1][0];

        // --------------------------------------------------------------
        // Stage 0: prefetch item ib+1, then wait for item ib's DMA.
        // TDM ops from one wave complete in order, so tensorcnt<=1 means
        // the older (current) transfer has landed in LDS.
        // --------------------------------------------------------------
#if HAVE_TDM
        if (__builtin_amdgcn_readfirstlane(wave) == 0) {
            const bool has_next = (ib + 1 < BPW) && (b + 1 < Btot);
            if (has_next) {
                issue_tdm((uint32_t)(uintptr_t)&xfc[(ib + 1) & 1][0],
                          x + (size_t)(b + 1) * ELEMS);
                __builtin_amdgcn_s_wait_tensorcnt(1);
            } else {
                __builtin_amdgcn_s_wait_tensorcnt(0);
            }
        }
#else
        {
            const float* xb = x + (size_t)b * ELEMS;
            for (int o = tid; o < ELEMS / 4; o += 256) {
                int hw = o >> 6, c4 = (o & 63) << 2;
                int h = hw / 9, w2 = hw - h * 9;
                int n = (h / 3) * 3 + (w2 / 3);
                int p = (h % 3) * 3 + (w2 % 3);
                float4 v = *(const float4*)(xb + hw * CCH + c4);
                *(float4*)(cur + (n * 9 + p) * CCH + c4) = v;
            }
        }
#endif
        __syncthreads();

        // --------------------------------------------------------------
        // Stage 1: convert A to bf16 in LDS (zero-pad rows 81..95)
        // --------------------------------------------------------------
        for (int i = tid; i < (ROWS_PAD * CCH) / 8; i += 256) {
            int r  = i >> 5;              // 32 chunks of 8 per row
            int c8 = (i & 31) << 3;
            v8bf hv;
            if (r < ROWS) {
                const float* src = cur + r * CCH + c8;
#pragma unroll
                for (int e = 0; e < 8; ++e) hv[e] = (bf16)src[e];
            } else {
#pragma unroll
                for (int e = 0; e < 8; ++e) hv[e] = (bf16)0.0f;
            }
            *(v8bf*)(abf + r * CCH + c8) = hv;
        }
        __syncthreads();

        // --------------------------------------------------------------
        // Stage 2: q = A(96x256 bf16) x B(256x128 bf16) + bias, f32 acc.
        // wave = n-tile; 6 m-tiles x 8 k-steps = 48 WMMAs per wave.
        // --------------------------------------------------------------
        v8f acc[6];
#pragma unroll
        for (int mt = 0; mt < 6; ++mt) {
            v8f c = { 0.f, 0.f, 0.f, 0.f, 0.f, 0.f, 0.f, 0.f };
            const bf16* arow = abf + (mt * 16 + lmod) * CCH;
#pragma unroll
            for (int kk = 0; kk < 8; ++kk) {
                const int col = kk * 32 + lhal * 16;
                union { v16bf v; u32x4 u[2]; } a;
                a.u[0] = *(const u32x4*)(arow + col);
                a.u[1] = *(const u32x4*)(arow + col + 8);
                c = __builtin_amdgcn_wmma_f32_16x16x32_bf16(
                        false, a.v, false, bfrag[kk], (short)0, c, false, false);
            }
            acc[mt] = c;
        }
        // scatter C to LDS q: element i of v8f = C[m = lhal*8 + i][n = lmod]
#pragma unroll
        for (int mt = 0; mt < 6; ++mt) {
#pragma unroll
            for (int i = 0; i < 8; ++i) {
                int r = mt * 16 + lhal * 8 + i;
                if (r < ROWS)
                    qld[r * HID + wave * 16 + lmod] = acc[mt][i] + bias;
            }
        }
        __syncthreads();

        // --------------------------------------------------------------
        // Stage 3: attn[n][m] = sum_p sum_h q[n*9+p][h] * q[m*9+p][h]
        // 729 (n,m,p) tasks, 128-length dots, LDS atomic reduce over p.
        // --------------------------------------------------------------
        if (tid < NTOK * NTOK) attn_l[tid] = 0.f;
        __syncthreads();
        for (int t = tid; t < 729; t += 256) {
            int p  = t / 81;
            int nm = t - p * 81;
            int n  = nm / 9, m = nm - n * 9;
            const float* qa = qld + (n * 9 + p) * HID;
            const float* qb = qld + (m * 9 + p) * HID;
            float s = 0.f;
#pragma unroll 8
            for (int h = 0; h < HID; h += 4) {
                float4 A4 = *(const float4*)(qa + h);
                float4 B4 = *(const float4*)(qb + h);
                s += A4.x * B4.x + A4.y * B4.y + A4.z * B4.z + A4.w * B4.w;
            }
            atomicAdd(&attn_l[nm], s);   // ds_add_f32
        }
        __syncthreads();

        // --------------------------------------------------------------
        // Stage 4: mask + softmax per row (9 threads)
        // --------------------------------------------------------------
        if (tid < NTOK) {
            const int   n     = tid;
            const int   match = match_vec[block_idx[b]];
            const float scale = 0.02946278f;   // (128*9)^-0.5
            float lg[NTOK];
            float mx = -1e30f;
#pragma unroll
            for (int m = 0; m < NTOK; ++m) {
                float v = attn_l[n * 9 + m] * scale;
                if (m == n && match != 1) v -= 100.f;
                lg[m] = v;
                mx = fmaxf(mx, v);
            }
            float sum = 0.f;
#pragma unroll
            for (int m = 0; m < NTOK; ++m) { float e = __expf(lg[m] - mx); lg[m] = e; sum += e; }
            float inv = 1.f / sum;
#pragma unroll
            for (int m = 0; m < NTOK; ++m) attn_l[n * 9 + m] = lg[m] * inv;
        }
        __syncthreads();

        // --------------------------------------------------------------
        // Stage 5: out[n][p][c] = sum_m attn[n][m] * cur[m*9+p][c],
        // written back coalesced with the inverse patch permutation.
        // --------------------------------------------------------------
        float* outb = out + (size_t)b * ELEMS;
        for (int o = tid; o < ELEMS / 4; o += 256) {
            int hw = o >> 6, c4 = (o & 63) << 2;
            int h = hw / 9, w2 = hw - h * 9;
            int n = (h / 3) * 3 + (w2 / 3);
            int p = (h % 3) * 3 + (w2 % 3);
            const float* arow = attn_l + n * 9;
            float4 a4 = { 0.f, 0.f, 0.f, 0.f };
#pragma unroll
            for (int m = 0; m < NTOK; ++m) {
                float  a = arow[m];
                float4 v = *(const float4*)(cur + (m * 9 + p) * CCH + c4);
                a4.x += a * v.x; a4.y += a * v.y; a4.z += a * v.z; a4.w += a * v.w;
            }
            *(float4*)(outb + hw * CCH + c4) = a4;
        }
        __syncthreads();   // this buffer is the next iteration's TDM dest
    }
}

extern "C" void kernel_launch(void* const* d_in, const int* in_sizes, int n_in,
                              void* d_out, int out_size, void* d_ws, size_t ws_size,
                              hipStream_t stream)
{
    const float* x         = (const float*)d_in[0];
    const float* W_fc      = (const float*)d_in[1];
    const float* b_fc      = (const float*)d_in[2];
    const int*   block_idx = (const int*)d_in[3];
    const int*   match_vec = (const int*)d_in[4];
    float*       out       = (float*)d_out;

    const int B    = in_sizes[3];           // batch = len(block_idx)
    const int grid = (B + BPW - 1) / BPW;

    attn_pix_kernel<<<dim3(grid), dim3(256), 0, stream>>>(
        x, W_fc, b_fc, block_idx, match_vec, out, B);
}
